// TransformerVariableSelectionNetwork_31714038513802
// MI455X (gfx1250) — matrix operand order repaired and checked by reading, not compiled, not run on hardware
//
#include <hip/hip_runtime.h>
#include <hip/hip_bf16.h>
#include <math.h>

#define HID 256
#define NV  20
#define NH  8
#define DH  32
#define BT  (32 * 512)
#define RT  16                 // token rows per workgroup
#define NBLK (BT / RT)         // 1024 workgroups
#define EPS 1e-5f

typedef __attribute__((ext_vector_type(16))) __bf16 v16bf;
typedef __attribute__((ext_vector_type(8)))  __bf16 v8bf;
typedef __attribute__((ext_vector_type(8)))  float  v8f;

// ---- bf16 helpers (bit-exact RNE) ----
__device__ __forceinline__ __bf16 f2bf(float f) {
    union { float f; unsigned u; } x; x.f = f;
    unsigned r = (x.u + 0x7FFFu + ((x.u >> 16) & 1u)) >> 16;
    unsigned short s = (unsigned short)r;
    return __builtin_bit_cast(__bf16, s);
}
__device__ __forceinline__ float bf2f(__bf16 b) {
    unsigned short s = __builtin_bit_cast(unsigned short, b);
    union { unsigned u; float f; } x; x.u = ((unsigned)s) << 16;
    return x.f;
}

// ---- WMMA fragment loaders (ISA 7.12.2, wave32) ----
// A (16x32): lane&15 = row M; lanes<16 hold K {0..7,16..23}, lanes>=16 {8..15,24..31}
// -> two aligned 16B chunk loads (ds_load_b128 from LDS)
__device__ __forceinline__ v16bf load_a(const __bf16* rowp, int lane) {
    int kb = (lane & 16) ? 8 : 0;
    v8bf lo = *(const v8bf*)(rowp + kb);
    v8bf hi = *(const v8bf*)(rowp + 16 + kb);
    v16bf a;
#pragma unroll
    for (int e = 0; e < 8; ++e) { a[e] = lo[e]; a[8 + e] = hi[e]; }
    return a;
}
// B (32x16): lane&15 = col N; lanes<16 hold K {0..7,16..23}, lanes>=16 {8..15,24..31}; strided gather
__device__ __forceinline__ v16bf load_b(const __bf16* p, int stride, int lane) {
    int kb = (lane & 16) ? 8 : 0;
    v16bf b;
#pragma unroll
    for (int e = 0; e < 8; ++e) { b[e] = p[(kb + e) * stride]; b[8 + e] = p[(16 + kb + e) * stride]; }
    return b;
}
__device__ __forceinline__ v8f wmma_bf(v16bf a, v16bf b, v8f c) {
    return __builtin_amdgcn_wmma_f32_16x16x32_bf16(false, a, false, b, (short)0, c, false, false);
}

// =====================================================================
// Kernel 1: fold query:  kqb[c][h(16)] = bf16( sum_d wk[c][h*32+d]*q[h*32+d] )
//           sb[h] = sum_d bk[h*32+d]*q[h*32+d],  q = cls@wq + bq
//           wwb[c][v(32)] = bf16(ww[c][v])  (zero-padded cols)
// =====================================================================
__global__ void vsn_prep(const float* __restrict__ cls,
                         const float* __restrict__ wq, const float* __restrict__ bq,
                         const float* __restrict__ wk, const float* __restrict__ bk,
                         const float* __restrict__ ww,
                         __bf16* __restrict__ kqb, float* __restrict__ sb,
                         __bf16* __restrict__ wwb) {
    __shared__ float qsh[HID];
    int t = threadIdx.x;
    float acc = bq[t];
    for (int c = 0; c < HID; ++c) acc += cls[c] * wq[c * HID + t];
    qsh[t] = acc;
    __syncthreads();
    int c = t;
#pragma unroll
    for (int h = 0; h < 16; ++h) {
        float s = 0.f;
        if (h < NH) {
#pragma unroll
            for (int d = 0; d < DH; ++d) s += wk[c * HID + h * DH + d] * qsh[h * DH + d];
        }
        kqb[c * 16 + h] = f2bf(s);
    }
#pragma unroll
    for (int j = 0; j < 32; ++j)
        wwb[c * 32 + j] = (j < NV) ? f2bf(ww[c * NV + j]) : f2bf(0.f);
    if (t < NH) {
        float s = 0.f;
        for (int d = 0; d < DH; ++d) s += bk[t * DH + d] * qsh[t * DH + d];
        sb[t] = s;
    }
}

// =====================================================================
// Kernel 2: convert weight matrices fp32 -> bf16 (L2-resident afterwards)
// =====================================================================
__global__ void vsn_convert(const float* __restrict__ wv, const float* __restrict__ wo,
                            const float* __restrict__ w1, const float* __restrict__ w2,
                            __bf16* __restrict__ wvb, __bf16* __restrict__ wob,
                            __bf16* __restrict__ w1b, __bf16* __restrict__ w2b) {
    int i = blockIdx.x * blockDim.x + threadIdx.x;
    int stride = gridDim.x * blockDim.x;
    for (int k = i; k < HID * HID; k += stride) { wvb[k] = f2bf(wv[k]); wob[k] = f2bf(wo[k]); }
    for (int k = i; k < HID * 4 * HID; k += stride) { w1b[k] = f2bf(w1[k]); w2b[k] = f2bf(w2[k]); }
}

// =====================================================================
// Kernel 3: fused VSN main — one WG (8 wave32) per 16 tokens
// =====================================================================
// LDS layout (bytes); all offsets 32B-aligned
#define LDS_XS    0        // bf16 [320][256]   = 163840
#define LDS_ATTN  163840   // f32  [16][8][20]  = 10240
#define LDS_YBF   174080   // bf16 [128][256]   = 65536   (reused as h2 [16][1024])
#define LDS_CTX   239616   // bf16 [16][256]    = 8192
#define LDS_ABUF  247808   // f32  [16][256]    = 16384   (attn_out, then h3)
#define LDS_H1    264192   // bf16 [16][256]    = 8192    (reused as processed bf16)
#define LDS_MU    272384   // f32  [16]
#define LDS_RS    272448   // f32  [16]
#define LDS_WLOG  272512   // f32  [16][20]     = 1280
#define LDS_TOTAL 273792

__global__ __launch_bounds__(256, 1)
void vsn_main(const float* __restrict__ x,
              const float* __restrict__ bv,  const float* __restrict__ bo,
              const float* __restrict__ ln1g, const float* __restrict__ ln1b,
              const float* __restrict__ b1,  const float* __restrict__ b2,
              const float* __restrict__ ln2g, const float* __restrict__ ln2b,
              const float* __restrict__ bw,
              const __bf16* __restrict__ kqb, const float* __restrict__ sbv,
              const __bf16* __restrict__ wwb,
              const __bf16* __restrict__ wvb, const __bf16* __restrict__ wob,
              const __bf16* __restrict__ w1b, const __bf16* __restrict__ w2b,
              float* __restrict__ out) {
    extern __shared__ char smem[];
    __bf16* xs   = (__bf16*)(smem + LDS_XS);
    float*  attn = (float*)(smem + LDS_ATTN);
    __bf16* ybf  = (__bf16*)(smem + LDS_YBF);
    __bf16* h2bf = (__bf16*)(smem + LDS_YBF);
    __bf16* ctxb = (__bf16*)(smem + LDS_CTX);
    float*  abuf = (float*)(smem + LDS_ABUF);
    __bf16* h1bf = (__bf16*)(smem + LDS_H1);
    __bf16* pbf  = (__bf16*)(smem + LDS_H1);
    float*  mu   = (float*)(smem + LDS_MU);
    float*  rs   = (float*)(smem + LDS_RS);
    float*  wlog = (float*)(smem + LDS_WLOG);

    const int tid  = threadIdx.x;
    const int lane = tid & 31;
    const int wave = tid >> 5;
    const int row0 = blockIdx.x * RT;
    const float scale = 0.17677669529663687f;   // 1/sqrt(32)
    const int n    = lane & 15;
    const int mb   = (lane & 16) ? 8 : 0;

    // ---- stage x tile (16 tokens x 20 vars x 256) as bf16 into LDS ----
    {
        const float4* xg = (const float4*)(x + (size_t)row0 * NV * HID);
        for (int i = tid; i < RT * NV * HID / 4; i += 256) {
            float4 f = xg[i];
            int e = i * 4;
            xs[e + 0] = f2bf(f.x); xs[e + 1] = f2bf(f.y);
            xs[e + 2] = f2bf(f.z); xs[e + 3] = f2bf(f.w);
        }
    }
    __syncthreads();

    // ---- scores via WMMA:  X[320,256] @ kqb[256,16]  (heads in cols 0..7) ----
    {
        const bool hvalid = n < NH;
        const float sbh = hvalid ? sbv[n] : 0.f;
        for (int t = wave; t < 20; t += 8) {          // 20 M-tiles of 16 rows
            v8f acc = {};
#pragma unroll
            for (int ks = 0; ks < 8; ++ks) {
                int k0 = ks * 32;
                v16bf a = load_a(&xs[(t * 16 + n) * HID + k0], lane);
                v16bf b = load_b(&kqb[k0 * 16 + n], 16, lane);
                acc = wmma_bf(a, b, acc);
            }
            if (hvalid) {
#pragma unroll
                for (int i = 0; i < 8; ++i) {
                    int grow = t * 16 + mb + i;        // row in [0,320) = m*NV+v
                    int m = grow / NV, v = grow - m * NV;
                    attn[(m * NH + n) * NV + v] = (acc[i] + sbh) * scale;
                }
            }
        }
    }
    __syncthreads();

    // ---- softmax over v ----
    if (tid < RT * NH) {
        float* p = &attn[tid * NV];
        float mx = p[0];
        for (int v = 1; v < NV; ++v) mx = fmaxf(mx, p[v]);
        float sum = 0.f;
        for (int v = 0; v < NV; ++v) { float e = __expf(p[v] - mx); p[v] = e; sum += e; }
        float inv = 1.f / sum;
        for (int v = 0; v < NV; ++v) p[v] *= inv;
    }
    __syncthreads();

    // ---- y[m,h,:] = sum_v attn * x_v   (bf16, WMMA A operand) ----
    for (int idx = tid; idx < RT * NH * HID; idx += 256) {
        int row = idx / HID, c = idx % HID;
        int m = row / NH, h = row % NH;
        const float* ap = &attn[(m * NH + h) * NV];
        float acc = 0.f;
#pragma unroll
        for (int v = 0; v < NV; ++v) acc += ap[v] * bf2f(xs[(m * NV + v) * HID + c]);
        ybf[row * HID + c] = f2bf(acc);
    }
    __syncthreads();

    // ---- ctx: head h = wave.  Y_h[16,256] @ wv[:,h*32..h*32+32) + bv ----
    {
        const int h = wave;                            // 8 waves = 8 heads
        const int n0a = h * DH, n0b = h * DH + 16;
        v8f acc0 = {}, acc1 = {};
#pragma unroll
        for (int ks = 0; ks < 8; ++ks) {
            int k0 = ks * 32;
            v16bf a = load_a(&ybf[((n)*NH + h) * HID + k0], lane);
            acc0 = wmma_bf(a, load_b(&wvb[k0 * HID + n0a + n], HID, lane), acc0);
            acc1 = wmma_bf(a, load_b(&wvb[k0 * HID + n0b + n], HID, lane), acc1);
        }
        float ba = bv[n0a + n], bb = bv[n0b + n];
#pragma unroll
        for (int i = 0; i < 8; ++i) {
            ctxb[(mb + i) * HID + n0a + n] = f2bf(acc0[i] + ba);
            ctxb[(mb + i) * HID + n0b + n] = f2bf(acc1[i] + bb);
        }
    }
    __syncthreads();

    // ---- attn_out = ctx @ wo + bo  (hoisted A, dual accumulators) ----
    {
        v16bf af[8];
#pragma unroll
        for (int ks = 0; ks < 8; ++ks) af[ks] = load_a(&ctxb[n * HID + ks * 32], lane);
        const int n0a = wave * 16, n0b = (wave + 8) * 16;
        v8f acc0 = {}, acc1 = {};
#pragma unroll
        for (int ks = 0; ks < 8; ++ks) {
            int k0 = ks * 32;
            acc0 = wmma_bf(af[ks], load_b(&wob[k0 * HID + n0a + n], HID, lane), acc0);
            acc1 = wmma_bf(af[ks], load_b(&wob[k0 * HID + n0b + n], HID, lane), acc1);
        }
        float ba = bo[n0a + n], bb = bo[n0b + n];
#pragma unroll
        for (int i = 0; i < 8; ++i) {
            abuf[(mb + i) * HID + n0a + n] = acc0[i] + ba;
            abuf[(mb + i) * HID + n0b + n] = acc1[i] + bb;
        }
    }
    __syncthreads();

    // ---- LayerNorm 1 -> h1 (bf16) ----
    if (tid < RT) {
        float s = 0.f;
        for (int c = 0; c < HID; ++c) s += abuf[tid * HID + c];
        float m_ = s / HID, v_ = 0.f;
        for (int c = 0; c < HID; ++c) { float d = abuf[tid * HID + c] - m_; v_ += d * d; }
        mu[tid] = m_; rs[tid] = rsqrtf(v_ / HID + EPS);
    }
    __syncthreads();
    for (int idx = tid; idx < RT * HID; idx += 256) {
        int m = idx / HID, c = idx % HID;
        h1bf[idx] = f2bf((abuf[idx] - mu[m]) * rs[m] * ln1g[c] + ln1b[c]);
    }
    __syncthreads();

    // ---- FF1: h1 @ w1 + b1 -> GELU -> h2 (hoisted A, paired n-tiles) ----
    {
        v16bf af[8];
#pragma unroll
        for (int ks = 0; ks < 8; ++ks) af[ks] = load_a(&h1bf[n * HID + ks * 32], lane);
#pragma unroll
        for (int p = 0; p < 4; ++p) {
            const int n0a = (wave + 16 * p) * 16, n0b = (wave + 8 + 16 * p) * 16;
            v8f acc0 = {}, acc1 = {};
#pragma unroll
            for (int ks = 0; ks < 8; ++ks) {
                int k0 = ks * 32;
                acc0 = wmma_bf(af[ks], load_b(&w1b[k0 * 4 * HID + n0a + n], 4 * HID, lane), acc0);
                acc1 = wmma_bf(af[ks], load_b(&w1b[k0 * 4 * HID + n0b + n], 4 * HID, lane), acc1);
            }
            float ba = b1[n0a + n], bb = b1[n0b + n];
#pragma unroll
            for (int i = 0; i < 8; ++i) {
                float v0 = acc0[i] + ba;
                float v1 = acc1[i] + bb;
                v0 = 0.5f * v0 * (1.0f + erff(v0 * 0.70710678118654752f));
                v1 = 0.5f * v1 * (1.0f + erff(v1 * 0.70710678118654752f));
                h2bf[(mb + i) * 4 * HID + n0a + n] = f2bf(v0);
                h2bf[(mb + i) * 4 * HID + n0b + n] = f2bf(v1);
            }
        }
    }
    __syncthreads();

    // ---- FF2: h2 @ w2 + b2 -> abuf (dual accumulators) ----
    {
        const int n0a = wave * 16, n0b = (wave + 8) * 16;
        v8f acc0 = {}, acc1 = {};
#pragma unroll
        for (int ks = 0; ks < 32; ++ks) {
            int k0 = ks * 32;
            v16bf a = load_a(&h2bf[n * 4 * HID + k0], lane);
            acc0 = wmma_bf(a, load_b(&w2b[k0 * HID + n0a + n], HID, lane), acc0);
            acc1 = wmma_bf(a, load_b(&w2b[k0 * HID + n0b + n], HID, lane), acc1);
        }
        float ba = b2[n0a + n], bb = b2[n0b + n];
#pragma unroll
        for (int i = 0; i < 8; ++i) {
            abuf[(mb + i) * HID + n0a + n] = acc0[i] + ba;
            abuf[(mb + i) * HID + n0b + n] = acc1[i] + bb;
        }
    }
    __syncthreads();

    // ---- LayerNorm 2 -> processed (bf16, into pbf) ----
    if (tid < RT) {
        float s = 0.f;
        for (int c = 0; c < HID; ++c) s += abuf[tid * HID + c];
        float m_ = s / HID, v_ = 0.f;
        for (int c = 0; c < HID; ++c) { float d = abuf[tid * HID + c] - m_; v_ += d * d; }
        mu[tid] = m_; rs[tid] = rsqrtf(v_ / HID + EPS);
    }
    __syncthreads();
    for (int idx = tid; idx < RT * HID; idx += 256) {
        int m = idx / HID, c = idx % HID;
        pbf[idx] = f2bf((abuf[idx] - mu[m]) * rs[m] * ln2g[c] + ln2b[c]);
    }
    __syncthreads();

    // ---- selection logits via WMMA: processed[16,256] @ wwb[256,32] + bw ----
    if (wave < 2) {
        const int n0 = wave * 16;
        const int v = n0 + n;
        v8f acc = {};
#pragma unroll
        for (int ks = 0; ks < 8; ++ks) {
            int k0 = ks * 32;
            v16bf a = load_a(&pbf[n * HID + k0], lane);
            acc = wmma_bf(a, load_b(&wwb[k0 * 32 + n0 + n], 32, lane), acc);
        }
        if (v < NV) {
            float bwv = bw[v];
#pragma unroll
            for (int i = 0; i < 8; ++i) wlog[(mb + i) * NV + v] = acc[i] + bwv;
        }
    }
    __syncthreads();
    if (tid < RT) {
        float* p = &wlog[tid * NV];
        float mx = p[0];
        for (int v = 1; v < NV; ++v) mx = fmaxf(mx, p[v]);
        float sum = 0.f;
        for (int v = 0; v < NV; ++v) { float e = __expf(p[v] - mx); p[v] = e; sum += e; }
        float inv = 1.f / sum;
        for (int v = 0; v < NV; ++v) p[v] *= inv;
    }
    __syncthreads();

    // ---- outputs: vs_output = sum_v w*x ; weights ----
    float* out_w = out + (size_t)BT * HID;
    for (int idx = tid; idx < RT * HID; idx += 256) {
        int m = idx / HID, c = idx % HID;
        const float* wp = &wlog[m * NV];
        float acc = 0.f;
#pragma unroll
        for (int v = 0; v < NV; ++v) acc += wp[v] * bf2f(xs[(m * NV + v) * HID + c]);
        out[(size_t)(row0 + m) * HID + c] = acc;
    }
    for (int idx = tid; idx < RT * NV; idx += 256) {
        int m = idx / NV, v = idx % NV;
        out_w[(size_t)(row0 + m) * NV + v] = wlog[m * NV + v];
    }
}

// =====================================================================
extern "C" void kernel_launch(void* const* d_in, const int* in_sizes, int n_in,
                              void* d_out, int out_size, void* d_ws, size_t ws_size,
                              hipStream_t stream) {
    const float* x    = (const float*)d_in[0];
    const float* cls  = (const float*)d_in[1];
    const float* wq   = (const float*)d_in[2];
    const float* bq   = (const float*)d_in[3];
    const float* wk   = (const float*)d_in[4];
    const float* bk   = (const float*)d_in[5];
    const float* wv   = (const float*)d_in[6];
    const float* bv   = (const float*)d_in[7];
    const float* wo   = (const float*)d_in[8];
    const float* bo   = (const float*)d_in[9];
    const float* ln1g = (const float*)d_in[10];
    const float* ln1b = (const float*)d_in[11];
    const float* w1   = (const float*)d_in[12];
    const float* b1   = (const float*)d_in[13];
    const float* w2   = (const float*)d_in[14];
    const float* b2   = (const float*)d_in[15];
    const float* ln2g = (const float*)d_in[16];
    const float* ln2b = (const float*)d_in[17];
    const float* ww   = (const float*)d_in[18];
    const float* bw   = (const float*)d_in[19];

    char* ws = (char*)d_ws;
    __bf16* kqb = (__bf16*)(ws + 0);                   // 8 KB  [256][16]
    float*  sb  = (float*)(ws + 8192);                 // 32 B
    __bf16* wwb = (__bf16*)(ws + 8320);                // 16 KB [256][32]
    __bf16* wvb = (__bf16*)(ws + 32768);               // 128 KB
    __bf16* wob = (__bf16*)(ws + 32768 + 131072);      // 128 KB
    __bf16* w1b = (__bf16*)(ws + 32768 + 262144);      // 512 KB
    __bf16* w2b = (__bf16*)(ws + 32768 + 786432);      // 512 KB

    vsn_prep<<<1, 256, 0, stream>>>(cls, wq, bq, wk, bk, ww, kqb, sb, wwb);
    vsn_convert<<<512, 256, 0, stream>>>(wv, wo, w1, w2, wvb, wob, w1b, w2b);
    vsn_main<<<NBLK, 256, LDS_TOTAL, stream>>>(
        x, bv, bo, ln1g, ln1b, b1, b2, ln2g, ln2b, bw,
        kqb, sb, wwb, wvb, wob, w1b, w2b, (float*)d_out);
}
